// E8RHTEmbedding_3212635537649
// MI455X (gfx1250) — matrix operand order repaired and checked by reading, not compiled, not run on hardware
//
#include <hip/hip_runtime.h>

// E8-codebook embedding dequant + 2048-pt Hadamard + SV scale on gfx1250.
// FHT(x) = H64 @ X @ H32 with X = x.reshape(64,32); one Sylvester butterfly
// level of H64 is folded into v_pk_add_f16 on the B fragments, so each of the
// 16 output tiles per token is a single full-K v_wmma_f32_16x16x32_f16.
// One wave32 per token; all LDS producer/consumer pairs are same-wave.

typedef __attribute__((ext_vector_type(16))) _Float16     v16h;
typedef __attribute__((ext_vector_type(8)))  _Float16     v8h;
typedef __attribute__((ext_vector_type(8)))  float        v8f;
typedef __attribute__((ext_vector_type(8)))  unsigned int v8u;

union HF { v8u u; v16h h; };   // build +/-1 f16 fragments as packed dwords

#define N_PAD 2048
#define QROW  256   // N_PAD / 8 codes per token per codebook

// Packed pair of Hadamard signs: halves (c_even, c_even+1) of row r.
// +1 -> 0x3C00, -1 -> 0xBC00 (sign bit xor).
__device__ __forceinline__ unsigned int hpair(unsigned r, unsigned c_even) {
  unsigned p0 = (unsigned)__builtin_popcount(r & c_even) & 1u;
  unsigned p1 = (unsigned)__builtin_popcount(r & (c_even | 1u)) & 1u;
  return 0x3C003C00u ^ (p0 << 15) ^ (p1 << 31);
}

__global__ __launch_bounds__(256, 1)
void e8rht_embed_kernel(const int* __restrict__ input_ids,
                        const unsigned short* __restrict__ Qidxs,
                        const unsigned short* __restrict__ Qidxs2,
                        const _Float16* __restrict__ codebook,
                        const _Float16* __restrict__ codebook2,
                        const _Float16* __restrict__ SV,
                        const float* __restrict__ Wscale_p,
                        const float* __restrict__ inv_resid_p,
                        _Float16* __restrict__ out,
                        int n_tok)
{
  // 8 waves/block, 4096 halves (8KB) per wave: [0..2047]=X (reused as OUT),
  // [2048..4095]=Y.  LDS ops from one wave execute in order -> no barriers.
  __shared__ _Float16 smem[8][4096];

  const int lane = threadIdx.x & 31;
  const int wave = threadIdx.x >> 5;
  const int tok  = blockIdx.x * 8 + wave;
  if (tok >= n_tok) return;                 // whole wave exits together

  _Float16* X = smem[wave];                 // 64 x 32 halves, row-major
  _Float16* Y = smem[wave] + 2048;          // 64 x 32 halves, row-major

  const float wsc = *Wscale_p;
  const float irs = *inv_resid_p;

  const int tid = input_ids[tok];
  const unsigned short* q1 = Qidxs  + (size_t)tid * QROW;
  const unsigned short* q2 = Qidxs2 + (size_t)tid * QROW;

  // ---- Stage 0: dequant into X = deq.reshape(64, 32), fp16 ----------------
  #pragma unroll
  for (int t = 0; t < 8; ++t) {
    const int g = lane + 32 * t;
    const unsigned int i1 = q1[g];          // int16 reinterpret == & 0xFFFF
    const unsigned int i2 = q2[g];
    v8h c1 = *(const v8h*)(codebook  + (size_t)i1 * 8);   // b128 gather
    v8h c2 = *(const v8h*)(codebook2 + (size_t)i2 * 8);
    v8h m;
    #pragma unroll
    for (int j = 0; j < 8; ++j)
      m[j] = (_Float16)(((float)c1[j] + irs * (float)c2[j]) * wsc);
    *(v8h*)(X + g * 8) = m;                 // ds_store_b128
  }

  // ---- Stage 1: Y = H64 * X, with H64 = H2 (x) H32' folded ----------------
  // B layout (32x16 f16): lane<16 -> N=lane, K=0..7,16..23; lane>=16 -> N=lane-16,
  // K=8..15,24..31 (2 per VGPR).  S/D = top +/- bottom 32 rows of X (one butterfly).
  v16h S[2], Dd[2];
  {
    const int n  = lane & 15;
    const int k0 = (lane < 16) ? 0 : 8;
    #pragma unroll
    for (int J = 0; J < 2; ++J) {
      v16h b0, b1;
      #pragma unroll
      for (int k = 0; k < 16; ++k) {
        b0[k] = X[(k0 + (k < 8 ? k : k + 8)) * 32      + 16 * J + n];   // rows 0..31
        b1[k] = X[(32 + k0 + (k < 8 ? k : k + 8)) * 32 + 16 * J + n];   // rows 32..63
      }
      S[J]  = b0 + b1;                      // v_pk_add_f16
      Dd[J] = b0 - b1;
    }
  }

  // Only two distinct analytic A fragments remain: rows 0..15 (G0) and
  // rows 16..31 (G1) of (-1)^pc(row & col), col<32.  A layout: lane<16 ->
  // M=lane, V0..3 -> K0..7, V4..7 -> K16..23; lane>=16 -> M=lane-16, +8 on K.
  HF G0, G1;
  {
    const int m  = lane & 15;
    const int kb = (lane < 16) ? 0 : 8;
    #pragma unroll
    for (int v = 0; v < 8; ++v) {
      const int k0 = ((v < 4) ? (2 * v) : (16 + 2 * (v - 4))) + kb; // even
      G0.u[v] = hpair((unsigned)m,        (unsigned)k0);
      G1.u[v] = hpair((unsigned)(16 + m), (unsigned)k0);
    }
  }

  #pragma unroll
  for (int I = 0; I < 4; ++I) {
    // Y tile rows 16I..16I+15:  I=0: G0*S, I=1: G1*S, I=2: G0*D, I=3: G1*D
    const v16h a = (I & 1) ? G1.h : G0.h;
    #pragma unroll
    for (int J = 0; J < 2; ++J) {
      v8f c = {};
      c = __builtin_amdgcn_wmma_f32_16x16x32_f16(false, a, false,
              (I < 2) ? S[J] : Dd[J], (short)0, c, false, false);
      // C/D layout: VGPR r -> (M=r, N=lane) / (M=r+8, N=lane-16)
      const int n  = lane & 15;
      const int mb = (lane < 16) ? 0 : 8;
      #pragma unroll
      for (int r = 0; r < 8; ++r)
        Y[(16 * I + mb + r) * 32 + 16 * J + n] = (_Float16)c[r];
    }
  }

  // ---- Stage 2: OUT = Y * H32  (8 WMMAs, K=32) ----------------------------
  // Bh[J][k][n] = H32[K][16J+n]; H32 = H2 (x) H16 => Bh1 = Bh0 with the sign
  // flipped on K>=16 rows, i.e. on the upper lane half only.
  HF Bh0, Bh1;
  {
    const int n   = lane & 15;
    const int kb2 = (lane < 16) ? 0 : 16;
    const unsigned sm = (lane < 16) ? 0u : 0x80008000u;
    #pragma unroll
    for (int v = 0; v < 8; ++v) {
      Bh0.u[v] = hpair((unsigned)n, (unsigned)(kb2 + 2 * v));
      Bh1.u[v] = Bh0.u[v] ^ sm;
    }
  }

  #pragma unroll
  for (int I = 0; I < 4; ++I) {
    // A = Y[16I..16I+15][0..31]; A layout makes each half-K contiguous:
    // two b128 LDS loads per lane.
    const int m    = lane & 15;
    const int k0   = (lane < 16) ? 0 : 16;
    const int base = (16 * I + m) * 32;
    v8h lo = *(const v8h*)(Y + base + k0);         // K = k0 .. k0+7
    v8h hi = *(const v8h*)(Y + base + 8 + k0);     // K = 8+k0 .. 8+k0+7
    v16h a;
    #pragma unroll
    for (int e = 0; e < 8; ++e) { a[e] = lo[e]; a[e + 8] = hi[e]; }

    #pragma unroll
    for (int J = 0; J < 2; ++J) {
      v8f c = {};
      c = __builtin_amdgcn_wmma_f32_16x16x32_f16(false, a, false,
              J ? Bh1.h : Bh0.h, (short)0, c, false, false);
      const int n  = lane & 15;
      const int mb = (lane < 16) ? 0 : 8;
      #pragma unroll
      for (int r = 0; r < 8; ++r)                   // OUT overwrites X region
        X[(16 * I + mb + r) * 32 + 16 * J + n] = (_Float16)c[r];
    }
  }

  // ---- Epilogue: fused SV scale + coalesced b128 store --------------------
  _Float16* o = out + (size_t)tok * N_PAD;
  #pragma unroll
  for (int t = 0; t < 8; ++t) {
    const int base = (t * 32 + lane) * 8;           // 16B-aligned, coalesced
    v8h v = *(const v8h*)(X + base);                // ds_load_b128
    v8h s = *(const v8h*)(SV + base);               // global b128, coalesced
    v8h r;
    #pragma unroll
    for (int j = 0; j < 8; ++j)
      r[j] = (_Float16)((float)v[j] * (float)s[j]);
    *(v8h*)(o + base) = r;                          // global_store_b128
  }
}

extern "C" void kernel_launch(void* const* d_in, const int* in_sizes, int n_in,
                              void* d_out, int out_size, void* d_ws, size_t ws_size,
                              hipStream_t stream) {
  const int*            ids = (const int*)d_in[0];
  const unsigned short* q1  = (const unsigned short*)d_in[1]; // int16 data
  const unsigned short* q2  = (const unsigned short*)d_in[2];
  const _Float16*       cb1 = (const _Float16*)d_in[3];       // fp16
  const _Float16*       cb2 = (const _Float16*)d_in[4];       // fp16
  const _Float16*       sv  = (const _Float16*)d_in[5];       // fp16
  const float*          ws  = (const float*)d_in[6];
  const float*          irs = (const float*)d_in[7];
  _Float16*             out = (_Float16*)d_out;               // fp16 output

  const int n_tok = in_sizes[0];                 // 8*2048 = 16384
  const int grid  = (n_tok + 7) / 8;             // 8 tokens (waves) per block
  e8rht_embed_kernel<<<grid, 256, 0, stream>>>(ids, q1, q2, cb1, cb2, sv,
                                               ws, irs, out, n_tok);
}
